// hierEncoder_8383776162074
// MI455X (gfx1250) — matrix operand order repaired and checked by reading, not compiled, not run on hardware
//
#include <hip/hip_runtime.h>
#include <hip/hip_bf16.h>
#include <math.h>

// ---------------- problem constants ----------------
#define E       512
#define G3E     1536          // 3*E
#define TSEQ    4096
#define VOCAB   25002
#define NWG     16            // workgroups cooperating per sequence in the recurrence
#define JCHUNK  32            // h elements owned per workgroup (16*32 = 512)

typedef __bf16 bf16_t;
typedef bf16_t v16bf __attribute__((ext_vector_type(16)));
typedef bf16_t v8bf  __attribute__((ext_vector_type(8)));
typedef float  v8f   __attribute__((ext_vector_type(8)));

typedef unsigned int u32x4 __attribute__((ext_vector_type(4)));
typedef int          i32x4 __attribute__((ext_vector_type(4)));
typedef int          i32x8 __attribute__((ext_vector_type(8)));

#if defined(__has_builtin)
#  if __has_builtin(__builtin_amdgcn_tensor_load_to_lds)
#    define HAVE_TDM 1
#    if __clang_major__ >= 23
#      define CDNA5_TDM_6ARG 1
#    endif
#  endif
#endif

__device__ __forceinline__ float sigmoidf_fast(float v) {
    return 1.0f / (1.0f + __expf(-v));
}

#if defined(HAVE_TDM)
// 2D TDM load: rows x cols f32 tile, row stride in elements, into LDS byte addr.
// Descriptor layout per CDNA5 ISA ch.8 (D# group0 128b, group1 256b; groups 2/3 zero => 2D).
__device__ __forceinline__ void tdm_load_2d_f32(unsigned lds_addr, const void* gptr,
                                                unsigned rows, unsigned cols,
                                                unsigned row_stride) {
    unsigned long long ga = (unsigned long long)(size_t)gptr;
    u32x4 g0;
    g0[0] = 1u;                                              // count=1, user mode
    g0[1] = lds_addr;                                        // lds_addr [63:32]
    g0[2] = (unsigned)(ga & 0xffffffffu);                    // global_addr lo
    g0[3] = (unsigned)((ga >> 32) & 0x01ffffffu) | (2u << 30); // global_addr hi, type=2
    i32x8 g1;
    g1[0] = (int)(2u << 16);                                 // mask=0, data_size=4B
    g1[1] = (int)((cols & 0xffffu) << 16);                   // tensor_dim0 lo16 @bit48
    g1[2] = (int)(((cols >> 16) & 0xffffu) | ((rows & 0xffffu) << 16)); // dim0 hi | dim1 lo
    g1[3] = (int)(((rows >> 16) & 0xffffu) | ((cols & 0xffffu) << 16)); // dim1 hi | tile_dim0
    g1[4] = (int)(rows & 0xffffu);                           // tile_dim1 | tile_dim2=0
    g1[5] = (int)row_stride;                                 // tensor_dim0_stride lo32
    g1[6] = 0;
    g1[7] = 0;                                               // tensor_dim1_stride=0 (2D)
    i32x4 gz = {0, 0, 0, 0};
#if defined(CDNA5_TDM_6ARG)
    i32x8 gz8 = {0, 0, 0, 0, 0, 0, 0, 0};
    __builtin_amdgcn_tensor_load_to_lds(g0, g1, gz, gz, gz8, 0);
#else
    __builtin_amdgcn_tensor_load_to_lds(g0, g1, gz, gz, 0);
#endif
}
#endif

// =====================================================================
// Kernel 0a: emb [V][512] f32 -> bf16 (one-time; gathered rows become pure
// b128 loads feeding WMMA with no per-iteration converts)
// =====================================================================
__global__ void k_prep_emb(const float* __restrict__ emb, bf16_t* __restrict__ embB) {
    int i4 = (blockIdx.x * 256 + threadIdx.x) * 4;           // VOCAB*E divisible by 4
    if (i4 >= VOCAB * E) return;
    float4 v = *(const float4*)(emb + i4);
    embB[i4 + 0] = (bf16_t)v.x;
    embB[i4 + 1] = (bf16_t)v.y;
    embB[i4 + 2] = (bf16_t)v.z;
    embB[i4 + 3] = (bf16_t)v.w;
}

// Kernel 0b: downconvert W_ih [1536][512] f32 -> WihT [1536][512] bf16 (same layout, no transpose)
__global__ void k_prep_wihT(const float* __restrict__ Wih, bf16_t* __restrict__ WihT) {
    int idx = blockIdx.x * blockDim.x + threadIdx.x;
    if (idx >= E * G3E) return;
    int k = idx / G3E;
    int c = idx - k * G3E;
    WihT[(size_t)c * E + k] = (bf16_t)Wih[(size_t)c * E + k];
}

// =====================================================================
// Kernel 1: gx[t][c] = sum_k embB[tok[t]][k] * WihT[c][k] + bih[c]
// One wave computes a 16x64 output strip: A reused across 4 B tiles,
// 4 x v_wmma_f32_16x16x32_bf16 per K-step, K=512.
// =====================================================================
__global__ void k_gemm_gx(const int*    __restrict__ toks,
                          const bf16_t* __restrict__ embB,
                          const bf16_t* __restrict__ WihT,
                          const float*  __restrict__ bih,
                          float*        __restrict__ gx,   // [T][1536]
                          int T) {
    const int ngroups = G3E / 64;                            // 24
    int wid  = blockIdx.x * 8 + (threadIdx.x >> 5);
    int lane = threadIdx.x & 31;
    int mtile = wid / ngroups;
    int ngrp  = wid - mtile * ngroups;
    if (mtile >= T / 16) return;
    int t0 = mtile * 16;
    int c0 = ngrp * 64;

    // A operand: lane<16 -> M=lane, K runs {0..7,16..23}; lane>=16 -> M=lane-16, +8
    int tok = toks[t0 + (lane & 15)];
    const bf16_t* arow = embB + (size_t)tok * E;
    int kb = (lane < 16) ? 0 : 8;

    int N = lane & 15;
    float bv0 = bih[c0 + N],      bv1 = bih[c0 + 16 + N];
    float bv2 = bih[c0 + 32 + N], bv3 = bih[c0 + 48 + N];

    v8f acc0 = {}, acc1 = {}, acc2 = {}, acc3 = {};
    for (int k0 = 0; k0 < E; k0 += 32) {
        v8bf lo = *(const v8bf*)(arow + k0 + kb);            // 16B aligned b128
        v8bf hi = *(const v8bf*)(arow + k0 + 16 + kb);
        v16bf a;
#pragma unroll
        for (int e = 0; e < 8; ++e) { a[e] = lo[e]; a[8 + e] = hi[e]; }

        // B operand: lane&15 = N column, K order same as A ({0..7,16..23} +8 for upper half)
        const bf16_t* bcol = WihT + (size_t)(c0 + N) * E + k0 + kb;
        v16bf b0, b1, b2, b3;
#pragma unroll
        for (int e = 0; e < 8; ++e) {
            b0[e] = bcol[e];              b0[8 + e] = bcol[16 + e];
            b1[e] = bcol[16 * E + e];     b1[8 + e] = bcol[16 * E + 16 + e];
            b2[e] = bcol[32 * E + e];     b2[8 + e] = bcol[32 * E + 16 + e];
            b3[e] = bcol[48 * E + e];     b3[8 + e] = bcol[48 * E + 16 + e];
        }

        acc0 = __builtin_amdgcn_wmma_f32_16x16x32_bf16(false, a, false, b0, (short)0, acc0, false, false);
        acc1 = __builtin_amdgcn_wmma_f32_16x16x32_bf16(false, a, false, b1, (short)0, acc1, false, false);
        acc2 = __builtin_amdgcn_wmma_f32_16x16x32_bf16(false, a, false, b2, (short)0, acc2, false, false);
        acc3 = __builtin_amdgcn_wmma_f32_16x16x32_bf16(false, a, false, b3, (short)0, acc3, false, false);
    }

    // C/D layout: VGPR r -> M = r + (lane<16?0:8), N = lane&15
    int mhi = (lane < 16) ? 0 : 8;
    float* gbase = gx + (size_t)t0 * G3E + c0 + N;
#pragma unroll
    for (int r = 0; r < 8; ++r) {
        size_t ro = (size_t)(r + mhi) * G3E;
        gbase[ro]      = acc0[r] + bv0;
        gbase[ro + 16] = acc1[r] + bv1;
        gbase[ro + 32] = acc2[r] + bv2;
        gbase[ro + 48] = acc3[r] + bv3;
    }
}

// =====================================================================
// Kernel 2: GRU1 recurrence.  32 blocks: blockIdx/16 = sequence, %16 = j-chunk.
// Each WG keeps its 96 W_hh rows ({j, 512+j, 1024+j}) in LDS as f32 (192 KB,
// staged by the Tensor Data Mover), exchanges the 512-float h through L2 each
// step, and synchronizes with a cluster barrier + atomic counter.
// =====================================================================
__global__ void k_gru1_recur(const float* __restrict__ Whh,
                             const float* __restrict__ bhh,
                             const float* __restrict__ gx_all,  // [2][T][1536]
                             float*       __restrict__ hping,   // [2][2][512]
                             unsigned*    __restrict__ bar,     // [2]
                             int T) {
    extern __shared__ char smem[];
    float* wrowF = (float*)smem;                  // [96][512] f32
    float* hsh   = wrowF + 96 * E;                // [512]
    float* ghsh  = hsh + E;                       // [96]

    const int tid  = threadIdx.x;
    const int wave = tid >> 5, lane = tid & 31;
    const int s     = blockIdx.x / NWG;
    const int g     = blockIdx.x - s * NWG;
    const int jbase = g * JCHUNK;

#if defined(HAVE_TDM)
    // Tensor Data Mover: three 2D descriptors, 32x512 f32 each (one per gate block)
    if (tid == 0) {
        unsigned lbase = (unsigned)(size_t)(void*)wrowF;     // low 32 bits = LDS offset
#pragma unroll
        for (int gate = 0; gate < 3; ++gate) {
            const float* src = Whh + ((size_t)gate * E + jbase) * E;
            tdm_load_2d_f32(lbase + (unsigned)gate * 32u * E * 4u, src, 32u, E, E);
        }
        __builtin_amdgcn_s_wait_tensorcnt(0);
    }
    __syncthreads();
#else
    for (int idx = tid; idx < 96 * E; idx += 256) {
        int r = idx >> 9;
        int k = idx & (E - 1);
        int gr = (r < 32) ? (jbase + r)
               : (r < 64) ? (E + jbase + (r - 32))
                          : (2 * E + jbase + (r - 64));
        wrowF[idx] = Whh[(size_t)gr * E + k];
    }
    __syncthreads();
#endif

    // per-lane gate biases for the owned j chunk
    float b_r = 0.f, b_z = 0.f, b_n = 0.f;
    if (tid < JCHUNK) {
        int j = jbase + tid;
        b_r = bhh[j];
        b_z = bhh[E + j];
        b_n = bhh[2 * E + j];
    }

    float* hbase = hping + (size_t)s * 2 * E;
    const float* gxs = gx_all + (size_t)s * T * G3E;

    for (int t = 0; t < T; ++t) {
        const float* hin  = hbase + ((t)     & 1) * E;
        float*       hout = hbase + ((t + 1) & 1) * E;

        // pull current h (written by all 16 WGs last step) with agent scope
        for (int i = tid; i < E; i += 256)
            hsh[i] = __hip_atomic_load(&hin[i], __ATOMIC_RELAXED,
                                       __HIP_MEMORY_SCOPE_AGENT);
        __syncthreads();

        // 96 dot products of length 512: one wave per row, float4 LDS reads,
        // lane covers 16 contiguous k, shuffle reduce.
        for (int rr = wave; rr < 96; rr += 8) {
            const float4* wr4 = (const float4*)(wrowF + rr * E);
            const float4* h4  = (const float4*)hsh;
            float a = 0.f;
#pragma unroll
            for (int q = 0; q < 4; ++q) {
                float4 w = wr4[lane * 4 + q];
                float4 h = h4[lane * 4 + q];
                a += w.x * h.x + w.y * h.y + w.z * h.z + w.w * h.w;
            }
#pragma unroll
            for (int off = 16; off > 0; off >>= 1)
                a += __shfl_xor(a, off, 32);
            if (lane == 0) ghsh[rr] = a;
        }
        __syncthreads();

        // elementwise gate math for the 32 owned h elements
        if (tid < JCHUNK) {
            int j = jbase + tid;
            const float* gxt = gxs + (size_t)t * G3E;
            float ghr = ghsh[tid]      + b_r;
            float ghz = ghsh[32 + tid] + b_z;
            float ghn = ghsh[64 + tid] + b_n;
            float r = sigmoidf_fast(gxt[j]     + ghr);
            float z = sigmoidf_fast(gxt[E + j] + ghz);
            float n = tanhf(gxt[2 * E + j] + r * ghn);
            float hn = (1.f - z) * n + z * hsh[j];
            __hip_atomic_store(&hout[j], hn, __ATOMIC_RELAXED,
                               __HIP_MEMORY_SCOPE_AGENT);
            if (t + 1 < T) {  // warm L2->WGP path for next step's gate row
                const float* nx = gxs + (size_t)(t + 1) * G3E;
                __builtin_prefetch(nx + j, 0, 1);
                __builtin_prefetch(nx + E + j, 0, 1);
                __builtin_prefetch(nx + 2 * E + j, 0, 1);
            }
        }
        __threadfence();
        __syncthreads();

        // inter-workgroup step barrier: cluster barrier (CDNA5 fast path,
        // S_NOP when not cluster-dispatched) + monotonic atomic counter.
        if (tid == 0) atomicAdd(&bar[s], 1u);
        __builtin_amdgcn_s_cluster_barrier();
        if (tid == 0) {
            unsigned target = (unsigned)(t + 1) * NWG;
            while (__hip_atomic_load(&bar[s], __ATOMIC_ACQUIRE,
                                     __HIP_MEMORY_SCOPE_AGENT) < target)
                __builtin_amdgcn_s_sleep(1);
        }
        __syncthreads();
    }
    // T=4096 even -> final hidden lands in slot 0 of each sequence
}

// =====================================================================
// Kernel 3: GRU2 over [hidden_x, hidden_y] + l1/relu + l2/relu + log_softmax
// =====================================================================
__global__ void k_gru2_head(const float* __restrict__ hping,   // [2][2][512]
                            const float* __restrict__ W2ih, const float* __restrict__ W2hh,
                            const float* __restrict__ b2ih, const float* __restrict__ b2hh,
                            const float* __restrict__ l1W,  const float* __restrict__ l1b,
                            const float* __restrict__ l2W,  const float* __restrict__ l2b,
                            float* __restrict__ out) {
    __shared__ float hin[E], hst[E], gxv[G3E], ghv[G3E], o1[128], o2r[2];
    const int tid = threadIdx.x, wave = tid >> 5, lane = tid & 31;

    for (int i = tid; i < E; i += 256) hst[i] = 0.f;
    __syncthreads();

    for (int stepi = 0; stepi < 2; ++stepi) {
        const float* hsrc = hping + (size_t)stepi * 2 * E;   // slot 0 of seq stepi
        for (int i = tid; i < E; i += 256) hin[i] = hsrc[i];
        __syncthreads();

        for (int r = wave; r < G3E; r += 8) {
            const float* wi = W2ih + (size_t)r * E;
            const float* wh = W2hh + (size_t)r * E;
            float ax = 0.f, ah = 0.f;
            for (int k = lane; k < E; k += 32) {
                ax += wi[k] * hin[k];
                ah += wh[k] * hst[k];
            }
#pragma unroll
            for (int off = 16; off > 0; off >>= 1) {
                ax += __shfl_xor(ax, off, 32);
                ah += __shfl_xor(ah, off, 32);
            }
            if (lane == 0) { gxv[r] = ax + b2ih[r]; ghv[r] = ah + b2hh[r]; }
        }
        __syncthreads();

        for (int j = tid; j < E; j += 256) {
            float r = sigmoidf_fast(gxv[j] + ghv[j]);
            float z = sigmoidf_fast(gxv[E + j] + ghv[E + j]);
            float n = tanhf(gxv[2 * E + j] + r * ghv[2 * E + j]);
            hst[j] = (1.f - z) * n + z * hst[j];
        }
        __syncthreads();
    }

    for (int r = wave; r < 128; r += 8) {
        const float* w = l1W + (size_t)r * E;
        float a = 0.f;
        for (int k = lane; k < E; k += 32) a += w[k] * hst[k];
#pragma unroll
        for (int off = 16; off > 0; off >>= 1) a += __shfl_xor(a, off, 32);
        if (lane == 0) o1[r] = fmaxf(a + l1b[r], 0.f);
    }
    __syncthreads();

    if (wave < 2) {
        const float* w = l2W + (size_t)wave * 128;
        float a = 0.f;
        for (int k = lane; k < 128; k += 32) a += w[k] * o1[k];
#pragma unroll
        for (int off = 16; off > 0; off >>= 1) a += __shfl_xor(a, off, 32);
        if (lane == 0) o2r[wave] = fmaxf(a + l2b[wave], 0.f);
    }
    __syncthreads();

    if (tid == 0) {
        float m   = fmaxf(o2r[0], o2r[1]);
        float lse = m + logf(__expf(o2r[0] - m) + __expf(o2r[1] - m));
        out[0] = o2r[0] - lse;
        out[1] = o2r[1] - lse;
    }
}

// =====================================================================
// Launch
// =====================================================================
extern "C" void kernel_launch(void* const* d_in, const int* in_sizes, int n_in,
                              void* d_out, int out_size, void* d_ws, size_t ws_size,
                              hipStream_t stream) {
    const int*   x      = (const int*)  d_in[0];
    const int*   y      = (const int*)  d_in[1];
    const float* emb    = (const float*)d_in[2];
    const float* g1_Wih = (const float*)d_in[3];
    const float* g1_Whh = (const float*)d_in[4];
    const float* g1_bih = (const float*)d_in[5];
    const float* g1_bhh = (const float*)d_in[6];
    const float* g2_Wih = (const float*)d_in[7];
    const float* g2_Whh = (const float*)d_in[8];
    const float* g2_bih = (const float*)d_in[9];
    const float* g2_bhh = (const float*)d_in[10];
    const float* l1_W   = (const float*)d_in[11];
    const float* l1_b   = (const float*)d_in[12];
    const float* l2_W   = (const float*)d_in[13];
    const float* l2_b   = (const float*)d_in[14];
    float* out = (float*)d_out;

    // workspace layout
    char* ws = (char*)d_ws;
    const size_t GX_BYTES  = (size_t)2 * TSEQ * G3E * sizeof(float);    // 50.3 MB
    const size_t EMB_BYTES = (size_t)VOCAB * E * sizeof(bf16_t);        // 25.6 MB
    const size_t WT_BYTES  = (size_t)E * G3E * sizeof(bf16_t);          // 1.5 MB
    const size_t HP_BYTES  = (size_t)2 * 2 * E * sizeof(float);         // 8 KB
    float*    gx    = (float*)ws;
    bf16_t*   embB  = (bf16_t*)(ws + GX_BYTES);
    bf16_t*   WihT  = (bf16_t*)(ws + GX_BYTES + EMB_BYTES);
    float*    hping = (float*)(ws + GX_BYTES + EMB_BYTES + WT_BYTES);
    unsigned* bar   = (unsigned*)(ws + GX_BYTES + EMB_BYTES + WT_BYTES + HP_BYTES);

    // allow >64KB dynamic LDS for the recurrence kernel (192 KB weights + vectors)
    const size_t lds = (size_t)96 * E * sizeof(float) + (E + 96) * sizeof(float);
    (void)hipFuncSetAttribute((const void*)k_gru1_recur,
                              hipFuncAttributeMaxDynamicSharedMemorySize, (int)lds);

    // zero h0 ping-pong buffers + barrier counters (graph-capture safe)
    hipMemsetAsync(hping, 0, HP_BYTES + 256, stream);

    // 0) one-time bf16 conversions
    k_prep_emb<<<(VOCAB * E / 4 + 255) / 256, 256, 0, stream>>>(emb, embB);
    k_prep_wihT<<<(E * G3E + 255) / 256, 256, 0, stream>>>(g1_Wih, WihT);

    // 1) WMMA input-gate GEMMs for both sequences (16x64 strip per wave)
    const int blocks = (TSEQ / 16) * (G3E / 64) / 8;   // 768
    k_gemm_gx<<<blocks, 256, 0, stream>>>(x, embB, WihT, g1_bih, gx, TSEQ);
    k_gemm_gx<<<blocks, 256, 0, stream>>>(y, embB, WihT, g1_bih,
                                          gx + (size_t)TSEQ * G3E, TSEQ);

    // 2) cooperative recurrence: 16 WGs per sequence, TDM-staged LDS W_hh slices
    k_gru1_recur<<<2 * NWG, 256, lds, stream>>>(g1_Whh, g1_bhh, gx, hping, bar, TSEQ);

    // 3) GRU2 + MLP head + log_softmax
    k_gru2_head<<<1, 256, 0, stream>>>(hping, g2_Wih, g2_Whh, g2_bih, g2_bhh,
                                       l1_W, l1_b, l2_W, l2_b, out);
}
